// GATGRUCell_36009005809881
// MI455X (gfx1250) — matrix operand order repaired and checked by reading, not compiled, not run on hardware
//
#include <hip/hip_runtime.h>
#include <hip/hip_bf16.h>

#define ALPHA 0.2f

typedef __attribute__((ext_vector_type(16))) __bf16 v16bf;
typedef __attribute__((ext_vector_type(8)))  __bf16 v8bf;
typedef __attribute__((ext_vector_type(8)))  float  v8f;

// ---------- helpers ----------
__device__ __forceinline__ __bf16 f2bf(float f) {
    unsigned u = __float_as_uint(f);
    unsigned r = u + 0x7FFFu + ((u >> 16) & 1u);   // round-to-nearest-even
    unsigned short h = (unsigned short)(r >> 16);
    return __builtin_bit_cast(__bf16, h);
}
__device__ __forceinline__ unsigned f2ord(float f) {
    unsigned u = __float_as_uint(f);
    return (u & 0x80000000u) ? ~u : (u | 0x80000000u);
}
__device__ __forceinline__ float ord2f(unsigned u) {
    unsigned v = (u & 0x80000000u) ? (u & 0x7FFFFFFFu) : ~u;
    return __uint_as_float(v);
}

// ---------- small elementwise kernels ----------
__global__ void k_f32_to_bf16(const float* __restrict__ in, __bf16* __restrict__ out, int n) {
    int i = blockIdx.x * blockDim.x + threadIdx.x;
    if (i < n) out[i] = f2bf(in[i]);
}
__global__ void k_fill_f32(float* __restrict__ p, float v, int n) {
    int i = blockIdx.x * blockDim.x + threadIdx.x;
    if (i < n) p[i] = v;
}
__global__ void k_fill_u32(unsigned* __restrict__ p, unsigned v, int n) {
    int i = blockIdx.x * blockDim.x + threadIdx.x;
    if (i < n) p[i] = v;
}
// W[H=8, F=256, D=32] -> Bt[C=256][K=256] bf16 with C = h*32+d, K = f
__global__ void k_repack_w(const float* __restrict__ W, __bf16* __restrict__ Bt) {
    int i = blockIdx.x * blockDim.x + threadIdx.x;   // 65536
    int c = i >> 8, f = i & 255;
    Bt[c * 256 + f] = f2bf(W[(c >> 5) * 8192 + f * 32 + (c & 31)]);
}
// es[n*8+h] = sum_d Wh[n*256 + h*32+d] * a[h*32+d]
__global__ void k_scores(const float* __restrict__ Wh, const float* __restrict__ a,
                         float* __restrict__ es, int Nn) {
    int i = blockIdx.x * blockDim.x + threadIdx.x;
    if (i >= Nn * 8) return;
    int n = i >> 3, h = i & 7;
    const float* wp = Wh + (size_t)n * 256 + h * 32;
    const float* ap = a + h * 32;
    float s = 0.f;
#pragma unroll
    for (int d = 0; d < 32; ++d) s += wp[d] * ap[d];
    es[i] = s;
}

// ---------- WMMA bf16 GEMM: C[M,N] = A[M,K=256] * B^T (B stored [N,K]) + bias[N] ----------
#define LDSTR 40   // 80B row stride: 16B aligned, conflict-spread
__global__ __launch_bounds__(256) void k_gemm_bf16_wmma(
    const __bf16* __restrict__ A, const __bf16* __restrict__ B,
    const float* __restrict__ bias, float* __restrict__ C,
    int M, int N, int K) {
    __shared__ __bf16 As[128 * LDSTR];
    __shared__ __bf16 Bs[128 * LDSTR];
    const int tid  = threadIdx.x;
    const int lane = tid & 31;
    const int wave = tid >> 5;
    const int wm = (wave >> 1) * 32;   // 0,32,64,96
    const int wn = (wave & 1) * 64;    // 0,64
    const int m0 = blockIdx.x * 128;
    const int n0 = blockIdx.y * 128;
    const int half = lane >> 4;        // 0/1
    const int l16  = lane & 15;

    v8f acc[2][4];
#pragma unroll
    for (int tm = 0; tm < 2; ++tm)
#pragma unroll
        for (int tn = 0; tn < 4; ++tn)
#pragma unroll
            for (int j = 0; j < 8; ++j) acc[tm][tn][j] = 0.f;

    const int lrow = tid >> 1;           // 0..127
    const int lhal = (tid & 1) * 16;     // 0/16 bf16 elems

    for (int k0 = 0; k0 < K; k0 += 32) {
        // stage A tile [128 x 32] (guard M) and B tile [128 x 32] (N multiple of 128)
        v8bf a0, a1;
#pragma unroll
        for (int j = 0; j < 8; ++j) { a0[j] = (__bf16)0.0f; a1[j] = (__bf16)0.0f; }
        int gm = m0 + lrow;
        if (gm < M) {
            const v8bf* src = (const v8bf*)(A + (size_t)gm * K + k0 + lhal);
            a0 = src[0]; a1 = src[1];
        }
        v8bf* adst = (v8bf*)(As + lrow * LDSTR + lhal);
        adst[0] = a0; adst[1] = a1;
        const v8bf* bsrc = (const v8bf*)(B + (size_t)(n0 + lrow) * K + k0 + lhal);
        v8bf* bdst = (v8bf*)(Bs + lrow * LDSTR + lhal);
        bdst[0] = bsrc[0]; bdst[1] = bsrc[1];
        __syncthreads();

        // A fragment (ISA 16-bit A 16x32): lanes 0-15 K=0..7 & 16..23, lanes 16-31 K=8..15 & 24..31
        v16bf afrag[2], bfrag[4];
#pragma unroll
        for (int tm = 0; tm < 2; ++tm) {
            const __bf16* p = As + (wm + tm * 16 + l16) * LDSTR + half * 8;
            v8bf lo = *(const v8bf*)p;
            v8bf hi = *(const v8bf*)(p + 16);
            afrag[tm] = __builtin_shufflevector(lo, hi, 0,1,2,3,4,5,6,7,8,9,10,11,12,13,14,15);
        }
        // B fragment: lanes 0-15 K=0..7 & 16..23, lanes 16-31 K=8..15 & 24..31 (same as A)
#pragma unroll
        for (int tn = 0; tn < 4; ++tn) {
            const __bf16* p = Bs + (wn + tn * 16 + l16) * LDSTR + half * 8;
            v8bf lo = *(const v8bf*)p;
            v8bf hi = *(const v8bf*)(p + 16);
            bfrag[tn] = __builtin_shufflevector(lo, hi, 0,1,2,3,4,5,6,7,8,9,10,11,12,13,14,15);
        }
#pragma unroll
        for (int tm = 0; tm < 2; ++tm)
#pragma unroll
            for (int tn = 0; tn < 4; ++tn)
                acc[tm][tn] = __builtin_amdgcn_wmma_f32_16x16x32_bf16(
                    false, afrag[tm], false, bfrag[tn], (short)0, acc[tm][tn], false, false);
        __syncthreads();
    }

    // epilogue: C/D layout m = r + 8*half, n = lane&15
#pragma unroll
    for (int tm = 0; tm < 2; ++tm)
#pragma unroll
        for (int tn = 0; tn < 4; ++tn) {
            int gn = n0 + wn + tn * 16 + l16;
            float bv = bias[gn];
#pragma unroll
            for (int r = 0; r < 8; ++r) {
                int gm = m0 + wm + tm * 16 + r + 8 * half;
                if (gm < M) C[(size_t)gm * N + gn] = acc[tm][tn][r] + bv;
            }
        }
}

// ---------- edge kernels ----------
__global__ void k_edge_max(const float* __restrict__ es, const float* __restrict__ ed,
                           const float* __restrict__ ab, const int* __restrict__ src,
                           const int* __restrict__ dst, unsigned* __restrict__ emax, int E) {
    int e = blockIdx.x * blockDim.x + threadIdx.x;
    if (e >= E) return;
    int s = src[e], d = dst[e];
#pragma unroll
    for (int h = 0; h < 8; ++h) {
        float v = es[s * 8 + h] + ed[d * 8 + h] + ab[h];
        v = v > 0.f ? v : ALPHA * v;
        atomicMax(&emax[d * 8 + h], f2ord(v));
    }
}
__global__ void k_edge_exp(const float* __restrict__ es, const float* __restrict__ ed,
                           const float* __restrict__ ab, const int* __restrict__ src,
                           const int* __restrict__ dst, const unsigned* __restrict__ emax,
                           float* __restrict__ denom, float* __restrict__ exbuf, int E) {
    int e = blockIdx.x * blockDim.x + threadIdx.x;
    if (e >= E) return;
    int s = src[e], d = dst[e];
#pragma unroll
    for (int h = 0; h < 8; ++h) {
        float v = es[s * 8 + h] + ed[d * 8 + h] + ab[h];
        v = v > 0.f ? v : ALPHA * v;
        float ex = __expf(v - ord2f(emax[d * 8 + h]));
        exbuf[(size_t)e * 8 + h] = ex;
        atomicAdd(&denom[d * 8 + h], ex);
    }
}
// one wave per edge; lane covers 8 contiguous channels (coalesced Wh reads)
__global__ __launch_bounds__(256) void k_edge_scatter(
    const float* __restrict__ exbuf, const float* __restrict__ denom,
    const float* __restrict__ Whs, const int* __restrict__ src,
    const int* __restrict__ dst, float* __restrict__ out, float coeff, int E) {
    int e = blockIdx.x * 8 + (threadIdx.x >> 5);
    if (e >= E) return;
    int lane = threadIdx.x & 31;
    int h = lane >> 2;                       // 8 channels per lane stay in one head
    int s = src[e], d = dst[e];
    float att = exbuf[(size_t)e * 8 + h] / denom[d * 8 + h] * coeff;
    const float* wp = Whs + (size_t)s * 256 + lane * 8;
    float* op = out + (size_t)d * 256 + lane * 8;
#pragma unroll
    for (int i = 0; i < 8; ++i) atomicAdd(&op[i], att * wp[i]);
}

// ---------- GRU elementwise ----------
__global__ void k_gru(const float* __restrict__ gi, const float* __restrict__ gh,
                      const float* __restrict__ hmix, float* __restrict__ out, int Nn) {
    int i = blockIdx.x * blockDim.x + threadIdx.x;
    if (i >= Nn * 256) return;
    int n = i >> 8, j = i & 255;
    size_t b = (size_t)n * 768;
    float ir = gi[b + j], iz = gi[b + 256 + j], inn = gi[b + 512 + j];
    float hr = gh[b + j], hz = gh[b + 256 + j], hn = gh[b + 512 + j];
    float r = 1.f / (1.f + __expf(-(ir + hr)));
    float z = 1.f / (1.f + __expf(-(iz + hz)));
    float nn = tanhf(inn + r * hn);
    out[i] = (1.f - z) * nn + z * hmix[i];
}

// ---------- host ----------
extern "C" void kernel_launch(void* const* d_in, const int* in_sizes, int n_in,
                              void* d_out, int out_size, void* d_ws, size_t ws_size,
                              hipStream_t stream) {
    const float* h_in      = (const float*)d_in[0];
    const float* hp_cnt    = (const float*)d_in[1];
    const float* hp_sup    = (const float*)d_in[2];
    const float* hp_dst    = (const float*)d_in[3];
    const float* W_gat     = (const float*)d_in[4];
    const float* b_gat     = (const float*)d_in[5];
    const float* as_gat    = (const float*)d_in[6];
    const float* ad_gat    = (const float*)d_in[7];
    const float* ab_gat    = (const float*)d_in[8];
    const float* W_x       = (const float*)d_in[9];
    const float* b_x       = (const float*)d_in[10];
    const float* as_x      = (const float*)d_in[11];
    const float* ad_x      = (const float*)d_in[12];
    const float* ab_x      = (const float*)d_in[13];
    const float* W_ih      = (const float*)d_in[14];
    const float* W_hh      = (const float*)d_in[15];
    const float* b_ih      = (const float*)d_in[16];
    const float* b_hh      = (const float*)d_in[17];
    const int* e_src       = (const int*)d_in[18];
    const int* e_dst       = (const int*)d_in[19];
    const int* ec_src      = (const int*)d_in[20];
    const int* ec_dst      = (const int*)d_in[21];
    const int* es_src      = (const int*)d_in[22];
    const int* es_dst      = (const int*)d_in[23];

    const int Nn = in_sizes[0] / 256;
    const int E  = in_sizes[18];
    float* out = (float*)d_out;

    char* ws = (char*)d_ws;
    size_t off = 0;
    auto alloc = [&](size_t bytes) { size_t o = off; off += (bytes + 255) & ~(size_t)255; return o; };
    __bf16* Abf   = (__bf16*)(ws + alloc((size_t)Nn * 256 * 2));
    __bf16* Wgbf  = (__bf16*)(ws + alloc(256 * 256 * 2));
    __bf16* Wxbf  = (__bf16*)(ws + alloc(256 * 256 * 2));
    __bf16* Wihbf = (__bf16*)(ws + alloc(768 * 256 * 2));
    __bf16* Whhbf = (__bf16*)(ws + alloc(768 * 256 * 2));
    float* slots  = (float*)(ws + alloc((size_t)Nn * 768 * 4));   // Wh1/Whc | Whd | Whs ; later gi
    float* Wh0 = slots;
    float* Whd = slots + (size_t)Nn * 256;
    float* Whs = slots + (size_t)Nn * 512;
    float* gi  = slots;
    float* xout  = (float*)(ws + alloc((size_t)Nn * 256 * 4));
    float* hmix  = (float*)(ws + alloc((size_t)Nn * 256 * 4));
    float* sc0   = (float*)(ws + alloc((size_t)Nn * 8 * 4));
    float* sc1   = (float*)(ws + alloc((size_t)Nn * 8 * 4));
    float* sc2   = (float*)(ws + alloc((size_t)Nn * 8 * 4));
    unsigned* emax = (unsigned*)(ws + alloc((size_t)Nn * 8 * 4));
    float* denom = (float*)(ws + alloc((size_t)Nn * 8 * 4));
    float* exbuf = (float*)(ws + alloc((size_t)E * 8 * 4));
    float* gh    = (float*)(ws + alloc((size_t)Nn * 768 * 4));
    (void)ws_size; (void)n_in; (void)out_size;

    const int gm = (Nn + 127) / 128;
    dim3 g256((Nn * 256 + 255) / 256), g8((Nn * 8 + 255) / 256), gE((E + 255) / 256);
    dim3 gEw((E + 7) / 8);
    const unsigned NEG_INF_ORD = 0x007FFFFFu;   // f2ord(-inf)

    // weight prep
    k_repack_w<<<256, 256, 0, stream>>>(W_gat, Wgbf);
    k_repack_w<<<256, 256, 0, stream>>>(W_x, Wxbf);
    k_f32_to_bf16<<<(768 * 256 + 255) / 256, 256, 0, stream>>>(W_ih, Wihbf, 768 * 256);
    k_f32_to_bf16<<<(768 * 256 + 255) / 256, 256, 0, stream>>>(W_hh, Whhbf, 768 * 256);

    // ---- GAT1 (intra-turn, src==dst projection) ----
    k_f32_to_bf16<<<g256, 256, 0, stream>>>(h_in, Abf, Nn * 256);
    k_gemm_bf16_wmma<<<dim3(gm, 2), 256, 0, stream>>>(Abf, Wgbf, b_gat, Wh0, Nn, 256, 256);
    k_scores<<<g8, 256, 0, stream>>>(Wh0, as_gat, sc0, Nn);
    k_scores<<<g8, 256, 0, stream>>>(Wh0, ad_gat, sc1, Nn);
    k_fill_f32<<<g256, 256, 0, stream>>>(xout, 0.f, Nn * 256);
    k_fill_u32<<<g8, 256, 0, stream>>>(emax, NEG_INF_ORD, Nn * 8);
    k_fill_f32<<<g8, 256, 0, stream>>>(denom, 0.f, Nn * 8);
    k_edge_max<<<gE, 256, 0, stream>>>(sc0, sc1, ab_gat, e_src, e_dst, emax, E);
    k_edge_exp<<<gE, 256, 0, stream>>>(sc0, sc1, ab_gat, e_src, e_dst, emax, denom, exbuf, E);
    k_edge_scatter<<<gEw, 256, 0, stream>>>(exbuf, denom, Wh0, e_src, e_dst, xout, 1.0f, E);

    // ---- shared projections for GAT2/GAT3 ----
    k_f32_to_bf16<<<g256, 256, 0, stream>>>(hp_cnt, Abf, Nn * 256);
    k_gemm_bf16_wmma<<<dim3(gm, 2), 256, 0, stream>>>(Abf, Wxbf, b_x, Wh0, Nn, 256, 256); // Whc
    k_scores<<<g8, 256, 0, stream>>>(Wh0, as_x, sc0, Nn);
    k_f32_to_bf16<<<g256, 256, 0, stream>>>(hp_dst, Abf, Nn * 256);
    k_gemm_bf16_wmma<<<dim3(gm, 2), 256, 0, stream>>>(Abf, Wxbf, b_x, Whd, Nn, 256, 256);
    k_scores<<<g8, 256, 0, stream>>>(Whd, ad_x, sc1, Nn);
    k_f32_to_bf16<<<g256, 256, 0, stream>>>(hp_sup, Abf, Nn * 256);
    k_gemm_bf16_wmma<<<dim3(gm, 2), 256, 0, stream>>>(Abf, Wxbf, b_x, Whs, Nn, 256, 256);
    k_scores<<<g8, 256, 0, stream>>>(Whs, as_x, sc2, Nn);
    k_fill_f32<<<g256, 256, 0, stream>>>(hmix, 0.f, Nn * 256);

    // ---- GAT2 (counter) -> 0.5 into hmix ----
    k_fill_u32<<<g8, 256, 0, stream>>>(emax, NEG_INF_ORD, Nn * 8);
    k_fill_f32<<<g8, 256, 0, stream>>>(denom, 0.f, Nn * 8);
    k_edge_max<<<gE, 256, 0, stream>>>(sc0, sc1, ab_x, ec_src, ec_dst, emax, E);
    k_edge_exp<<<gE, 256, 0, stream>>>(sc0, sc1, ab_x, ec_src, ec_dst, emax, denom, exbuf, E);
    k_edge_scatter<<<gEw, 256, 0, stream>>>(exbuf, denom, Wh0, ec_src, ec_dst, hmix, 0.5f, E);

    // ---- GAT3 (support) -> 0.5 into hmix ----
    k_fill_u32<<<g8, 256, 0, stream>>>(emax, NEG_INF_ORD, Nn * 8);
    k_fill_f32<<<g8, 256, 0, stream>>>(denom, 0.f, Nn * 8);
    k_edge_max<<<gE, 256, 0, stream>>>(sc2, sc1, ab_x, es_src, es_dst, emax, E);
    k_edge_exp<<<gE, 256, 0, stream>>>(sc2, sc1, ab_x, es_src, es_dst, emax, denom, exbuf, E);
    k_edge_scatter<<<gEw, 256, 0, stream>>>(exbuf, denom, Whs, es_src, es_dst, hmix, 0.5f, E);

    // ---- GRU: gi = x@W_ih^T + b_ih ; gh = hmix@W_hh^T + b_hh ----
    k_f32_to_bf16<<<g256, 256, 0, stream>>>(xout, Abf, Nn * 256);
    k_gemm_bf16_wmma<<<dim3(gm, 6), 256, 0, stream>>>(Abf, Wihbf, b_ih, gi, Nn, 768, 256);
    k_f32_to_bf16<<<g256, 256, 0, stream>>>(hmix, Abf, Nn * 256);
    k_gemm_bf16_wmma<<<dim3(gm, 6), 256, 0, stream>>>(Abf, Whhbf, b_hh, gh, Nn, 768, 256);
    k_gru<<<g256, 256, 0, stream>>>(gi, gh, hmix, out, Nn);
}
